// acLSTM_41059887350465
// MI455X (gfx1250) — matrix operand order repaired and checked by reading, not compiled, not run on hardware
//
#include <hip/hip_runtime.h>
#include <hip/hip_bf16.h>
#include <stdint.h>

// ---------------- problem constants ----------------
#define B_    256
#define T_    256
#define C_    171
#define H_    1024
#define G_    4096          // 4*H
#define CPAD  192           // C padded to multiple of 32
#define KPAD1 (CPAD + H_)   // 1216 : layer-1 concatenated K
#define KPAD23 (2 * H_)     // 2048 : layers 2/3 concatenated K
#define NDEC  176           // 171 padded to multiple of 16

typedef __bf16 bf16;
typedef __bf16 v16bf __attribute__((ext_vector_type(16)));
typedef __bf16 v8bf  __attribute__((ext_vector_type(8)));
typedef float  v8f   __attribute__((ext_vector_type(8)));

__device__ __forceinline__ float sigmoidf_(float x) { return 1.0f / (1.0f + __expf(-x)); }

__device__ __forceinline__ v16bf cat8(v8bf lo, v8bf hi) {
  return __builtin_shufflevector(lo, hi, 0,1,2,3,4,5,6,7,8,9,10,11,12,13,14,15);
}

// CDNA5 async memory->LDS copy (16B per lane), tracked by ASYNCcnt.
// Generic pointers to __shared__ carry the LDS byte offset in their low 32 bits.
__device__ __forceinline__ void async_copy_16B(unsigned lds_addr, const bf16* gsrc) {
  asm volatile("global_load_async_to_lds_b128 %0, %1, off"
               :: "v"(lds_addr), "v"(gsrc) : "memory");
}

// ---------------- init / conversion kernels ----------------
__global__ void k_zero(uint32_t* p, int n) {
  int i = blockIdx.x * blockDim.x + threadIdx.x;
  if (i < n) p[i] = 0u;
}

// Wcat[n][k] = k<Kin ? W_ih[n][k] : (k<KinPad ? 0 : W_hh[n][k-KinPad]),  bf16
__global__ void k_wcat(const float* __restrict__ Wih, const float* __restrict__ Whh,
                       bf16* __restrict__ dst, int Kin, int KinPad, int Kpad, int total) {
  int i = blockIdx.x * blockDim.x + threadIdx.x;
  if (i >= total) return;
  int n = i / Kpad, k = i - n * Kpad;
  float v = 0.0f;
  if (k < Kin)          v = Wih[(size_t)n * Kin + k];
  else if (k >= KinPad) v = Whh[(size_t)n * H_ + (k - KinPad)];
  dst[i] = (bf16)v;
}

__global__ void k_wdec(const float* __restrict__ W, bf16* __restrict__ dst, int total) {
  int i = blockIdx.x * blockDim.x + threadIdx.x;
  if (i >= total) return;
  int n = i / H_, k = i - n * H_;
  dst[i] = (bf16)((n < C_) ? W[(size_t)n * H_ + k] : 0.0f);
}

__global__ void k_bias(const float* __restrict__ bih, const float* __restrict__ bhh,
                       float* __restrict__ dst, int n) {
  int i = blockIdx.x * blockDim.x + threadIdx.x;
  if (i < n) dst[i] = bih[i] + bhh[i];
}

__global__ void k_bdec(const float* __restrict__ b, float* __restrict__ dst) {
  int i = blockIdx.x * blockDim.x + threadIdx.x;
  if (i < NDEC) dst[i] = (i < C_) ? b[i] : 0.0f;
}

// scheduled-sampling input select: overwrite x-slab with seq[:,t,:] iff ground-truth step
__global__ void k_prep(const float* __restrict__ seq, const int* __restrict__ cond,
                       const int* __restrict__ gt, bf16* __restrict__ x1, int t) {
  int i = blockIdx.x * blockDim.x + threadIdx.x;
  if (i >= B_ * C_) return;
  int per = gt[0] + cond[0];
  bool use_gt = (per <= 0) ? true : ((t % per) < gt[0]);
  if (!use_gt) return;  // keep decoder-fed value (or zeros at t==0)
  int b = i / C_, n = i - b * C_;
  x1[(size_t)b * CPAD + n] = (bf16)seq[((size_t)b * T_ + t) * C_ + n];
}

// ---------------- fused LSTM layer: gates GEMM + cell update ----------------
// grid = (B/64, H/32), block = 256 (8 wave32 waves: 4 M-waves x 2 J-waves).
// Each wave computes 16x16 tiles of all four gates (rows j, H+j, 2H+j, 3H+j of Wcat),
// then does the elementwise LSTM update in-register. Tiles are staged into
// double-buffered LDS with global_load_async_to_lds_b128 (ASYNCcnt-tracked) so the
// L2-resident weight stream overlaps the WMMA pipeline.
__global__ __launch_bounds__(256)
void k_lstm_layer(const bf16* __restrict__ Ax, int ldAx, int kInChunks,
                  const bf16* __restrict__ Ah,
                  const bf16* __restrict__ Wcat, int Kpad,
                  const float* __restrict__ biasCat,
                  float* __restrict__ cSt, bf16* __restrict__ hOut) {
  __shared__ __align__(16) bf16 ldsA[2][64 * 32];   // 2 x 4 KB
  __shared__ __align__(16) bf16 ldsB[2][128 * 32];  // 2 x 8 KB

  const int tid   = threadIdx.x;
  const int wave  = tid >> 5, lane = tid & 31;
  const int mwave = wave & 3, jwave = wave >> 2;
  const int mBase = blockIdx.x * 64;
  const int j0    = blockIdx.y * 32;
  const int laneHi = lane >> 4, lm = lane & 15;
  const int kChunks = Kpad / 32;

  // cooperative staging indices
  const int aRow = tid >> 2;            // 64 rows, 4 threads/row, 8 elems (16B) each
  const int aCol = (tid & 3) * 8;
  const int bRow = tid >> 1;            // 128 rows, 2 threads/row, 16 elems each
  const int bCol = (tid & 1) * 16;
  const int grp  = bRow >> 4;           // 0..7 : (jwave', gate)
  const int gate = grp & 3, jw = grp >> 2;
  const int wRowG = gate * H_ + j0 + jw * 16 + (bRow & 15);

  // LDS byte addresses of this thread's staging slots (per buffer)
  const unsigned ldsAAddr[2] = {
    (unsigned)(uintptr_t)&ldsA[0][aRow * 32 + aCol],
    (unsigned)(uintptr_t)&ldsA[1][aRow * 32 + aCol] };
  const unsigned ldsBAddr[2] = {
    (unsigned)(uintptr_t)&ldsB[0][bRow * 32 + bCol],
    (unsigned)(uintptr_t)&ldsB[1][bRow * 32 + bCol] };

  // branch-free A source: two base pointers, second pre-biased by -kInChunks*32
  const bf16* aBase0 = Ax + (size_t)(mBase + aRow) * ldAx + aCol;
  const bf16* aBase1 = Ah + (size_t)(mBase + aRow) * H_ + aCol - (size_t)kInChunks * 32;
  const bf16* wBase  = Wcat + (size_t)wRowG * Kpad + bCol;

  v8f acc0 = {}, acc1 = {}, acc2 = {}, acc3 = {};

  // prologue: stage chunk 0 into buffer 0  (3 async ops / thread / chunk)
  {
    const bf16* a = (0 < kInChunks ? aBase0 : aBase1);
    async_copy_16B(ldsAAddr[0], a);
    async_copy_16B(ldsBAddr[0], wBase);
    async_copy_16B(ldsBAddr[0] + 16u, wBase + 8);
  }

  for (int kk = 0; kk < kChunks; ++kk) {
    const int buf = kk & 1;
    if (kk + 1 < kChunks) {
      // stage next chunk into the other buffer (safe: all waves passed the
      // end-of-iteration barrier of kk-1, so nobody still reads buf^1)
      const int kn = kk + 1;
      const bf16* a = ((kn < kInChunks) ? aBase0 : aBase1) + (size_t)kn * 32;
      const bf16* w = wBase + (size_t)kn * 32;
      async_copy_16B(ldsAAddr[buf ^ 1], a);
      async_copy_16B(ldsBAddr[buf ^ 1], w);
      async_copy_16B(ldsBAddr[buf ^ 1] + 16u, w + 8);
      // retire current chunk's 3 ops; leave the 3 just issued in flight
      asm volatile("s_wait_asynccnt 0x3" ::: "memory");
    } else {
      asm volatile("s_wait_asynccnt 0x0" ::: "memory");
    }
    __syncthreads();

    const bf16* LA = ldsA[buf];
    const bf16* LB = ldsB[buf];

    // A fragment (16x32 bf16; lanes<16: K 0-7/16-23, lanes>=16: K 8-15/24-31)
    const int aOff = (mwave * 16 + lm) * 32 + laneHi * 8;
    v16bf aF = cat8(*(const v8bf*)&LA[aOff], *(const v8bf*)&LA[aOff + 16]);

    // B fragments (32x16 bf16; lane = column n, lanes<16 hold K 0-7/16-23, lanes>=16 K 8-15/24-31)
    const int bB = (jwave * 4) * 16;
    const int bo0 = ((bB +  0) + lm) * 32 + laneHi * 8;
    const int bo1 = ((bB + 16) + lm) * 32 + laneHi * 8;
    const int bo2 = ((bB + 32) + lm) * 32 + laneHi * 8;
    const int bo3 = ((bB + 48) + lm) * 32 + laneHi * 8;
    v16bf bFi = cat8(*(const v8bf*)&LB[bo0], *(const v8bf*)&LB[bo0 + 16]);
    v16bf bFf = cat8(*(const v8bf*)&LB[bo1], *(const v8bf*)&LB[bo1 + 16]);
    v16bf bFg = cat8(*(const v8bf*)&LB[bo2], *(const v8bf*)&LB[bo2 + 16]);
    v16bf bFo = cat8(*(const v8bf*)&LB[bo3], *(const v8bf*)&LB[bo3 + 16]);

    acc0 = __builtin_amdgcn_wmma_f32_16x16x32_bf16(false, aF, false, bFi, (short)0, acc0, false, false);
    acc1 = __builtin_amdgcn_wmma_f32_16x16x32_bf16(false, aF, false, bFf, (short)0, acc1, false, false);
    acc2 = __builtin_amdgcn_wmma_f32_16x16x32_bf16(false, aF, false, bFg, (short)0, acc2, false, false);
    acc3 = __builtin_amdgcn_wmma_f32_16x16x32_bf16(false, aF, false, bFo, (short)0, acc3, false, false);

    __syncthreads();
  }

  // fused elementwise LSTM cell update (C layout: VGPR r -> row r (+8 for hi lanes), lane -> col)
  const int jg = j0 + jwave * 16 + lm;
  const float bi = biasCat[jg];
  const float bff = biasCat[H_ + jg];
  const float bg = biasCat[2 * H_ + jg];
  const float bo = biasCat[3 * H_ + jg];
  const int bRow0 = mBase + mwave * 16 + laneHi * 8;
#pragma unroll
  for (int r = 0; r < 8; ++r) {
    const int b = bRow0 + r;
    const float iv = sigmoidf_(acc0[r] + bi);
    const float fv = sigmoidf_(acc1[r] + bff);
    const float gv = tanhf(acc2[r] + bg);
    const float ov = sigmoidf_(acc3[r] + bo);
    const size_t idx = (size_t)b * H_ + jg;
    const float cn = fv * cSt[idx] + iv * gv;
    cSt[idx] = cn;
    hOut[idx] = (bf16)(ov * tanhf(cn));
  }
}

// ---------------- decoder: y = h3 @ Wdec^T + b ----------------
// 176 wave-tiles (16 M x 11 N), 8 waves/block -> 22 blocks. Direct global frag loads.
__global__ __launch_bounds__(256)
void k_decoder(const bf16* __restrict__ h3, const bf16* __restrict__ Wd,
               const float* __restrict__ bd, float* __restrict__ out,
               bf16* __restrict__ x1, int t) {
  const int tid = threadIdx.x, wave = tid >> 5, lane = tid & 31;
  const int wg = blockIdx.x * 8 + wave;          // 0..175
  const int mTile = wg / 11, nTile = wg - mTile * 11;
  const int mBase = mTile * 16, n0 = nTile * 16;
  const int laneHi = lane >> 4, lm = lane & 15;

  v8f acc = {};
  const bf16* aRow = h3 + (size_t)(mBase + lm) * H_ + laneHi * 8;
  const bf16* bRow = Wd + (size_t)(n0 + lm) * H_ + laneHi * 8;
#pragma unroll 4
  for (int kk = 0; kk < H_ / 32; ++kk) {
    v16bf aF = cat8(*(const v8bf*)(aRow + kk * 32), *(const v8bf*)(aRow + kk * 32 + 16));
    v16bf bF = cat8(*(const v8bf*)(bRow + kk * 32), *(const v8bf*)(bRow + kk * 32 + 16));
    acc = __builtin_amdgcn_wmma_f32_16x16x32_bf16(false, aF, false, bF, (short)0, acc, false, false);
  }

  const int n = n0 + lm;
  if (n < C_) {
    const float bn = bd[n];
#pragma unroll
    for (int r = 0; r < 8; ++r) {
      const int b = mBase + laneHi * 8 + r;
      const float y = acc[r] + bn;
      out[((size_t)b * T_ + t) * C_ + n] = y;          // fp32 result
      x1[(size_t)b * CPAD + n] = (bf16)y;              // self-feed path (bf16)
    }
  }
}

// ---------------- host launcher ----------------
extern "C" void kernel_launch(void* const* d_in, const int* in_sizes, int n_in,
                              void* d_out, int out_size, void* d_ws, size_t ws_size,
                              hipStream_t stream) {
  (void)in_sizes; (void)n_in; (void)out_size; (void)ws_size;

  const float* seq   = (const float*)d_in[0];
  const float* W_ih1 = (const float*)d_in[1];
  const float* W_hh1 = (const float*)d_in[2];
  const float* b_ih1 = (const float*)d_in[3];
  const float* b_hh1 = (const float*)d_in[4];
  const float* W_ih2 = (const float*)d_in[5];
  const float* W_hh2 = (const float*)d_in[6];
  const float* b_ih2 = (const float*)d_in[7];
  const float* b_hh2 = (const float*)d_in[8];
  const float* W_ih3 = (const float*)d_in[9];
  const float* W_hh3 = (const float*)d_in[10];
  const float* b_ih3 = (const float*)d_in[11];
  const float* b_hh3 = (const float*)d_in[12];
  const float* W_dec = (const float*)d_in[13];
  const float* b_dec = (const float*)d_in[14];
  const int*   cond  = (const int*)d_in[15];
  const int*   gt    = (const int*)d_in[16];
  float* out = (float*)d_out;

  // workspace carve-up (256B aligned)
  char* ws = (char*)d_ws;
  size_t off = 0;
  auto carve = [&](size_t bytes) -> char* {
    char* p = ws + off;
    off += (bytes + 255) & ~(size_t)255;
    return p;
  };
  // zero-initialized state region first (contiguous)
  float* c1 = (float*)carve((size_t)B_ * H_ * 4);
  float* c2 = (float*)carve((size_t)B_ * H_ * 4);
  float* c3 = (float*)carve((size_t)B_ * H_ * 4);
  bf16* h1b[2] = { (bf16*)carve((size_t)B_ * H_ * 2), (bf16*)carve((size_t)B_ * H_ * 2) };
  bf16* h2b[2] = { (bf16*)carve((size_t)B_ * H_ * 2), (bf16*)carve((size_t)B_ * H_ * 2) };
  bf16* h3b[2] = { (bf16*)carve((size_t)B_ * H_ * 2), (bf16*)carve((size_t)B_ * H_ * 2) };
  bf16* x1 = (bf16*)carve((size_t)B_ * CPAD * 2);
  const size_t stateBytes = off;  // multiple of 256

  float* bias1 = (float*)carve(G_ * 4);
  float* bias2 = (float*)carve(G_ * 4);
  float* bias3 = (float*)carve(G_ * 4);
  float* bdec  = (float*)carve(NDEC * 4);
  bf16* Wcat1 = (bf16*)carve((size_t)G_ * KPAD1 * 2);
  bf16* Wcat2 = (bf16*)carve((size_t)G_ * KPAD23 * 2);
  bf16* Wcat3 = (bf16*)carve((size_t)G_ * KPAD23 * 2);
  bf16* WdecB = (bf16*)carve((size_t)NDEC * H_ * 2);

  // --- init: zero state, convert weights/biases to bf16 layouts ---
  {
    int words = (int)(stateBytes / 4);
    k_zero<<<(words + 255) / 256, 256, 0, stream>>>((uint32_t*)d_ws, words);
  }
  {
    int tot1 = G_ * KPAD1;
    k_wcat<<<(tot1 + 255) / 256, 256, 0, stream>>>(W_ih1, W_hh1, Wcat1, C_, CPAD, KPAD1, tot1);
    int tot2 = G_ * KPAD23;
    k_wcat<<<(tot2 + 255) / 256, 256, 0, stream>>>(W_ih2, W_hh2, Wcat2, H_, H_, KPAD23, tot2);
    k_wcat<<<(tot2 + 255) / 256, 256, 0, stream>>>(W_ih3, W_hh3, Wcat3, H_, H_, KPAD23, tot2);
    int totd = NDEC * H_;
    k_wdec<<<(totd + 255) / 256, 256, 0, stream>>>(W_dec, WdecB, totd);
    k_bias<<<(G_ + 255) / 256, 256, 0, stream>>>(b_ih1, b_hh1, bias1, G_);
    k_bias<<<(G_ + 255) / 256, 256, 0, stream>>>(b_ih2, b_hh2, bias2, G_);
    k_bias<<<(G_ + 255) / 256, 256, 0, stream>>>(b_ih3, b_hh3, bias3, G_);
    k_bdec<<<1, 256, 0, stream>>>(b_dec, bdec);
  }

  // --- sequential timesteps ---
  const dim3 lgrid(B_ / 64, H_ / 32);  // 4 x 32 workgroups
  for (int t = 0; t < T_; ++t) {
    const int p = t & 1, q = (t + 1) & 1;
    k_prep<<<(B_ * C_ + 255) / 256, 256, 0, stream>>>(seq, cond, gt, x1, t);
    // layer 1: A = [x (padded 192) | h1_prev]
    k_lstm_layer<<<lgrid, 256, 0, stream>>>(x1, CPAD, CPAD / 32, h1b[p],
                                            Wcat1, KPAD1, bias1, c1, h1b[q]);
    // layer 2: A = [h1_new | h2_prev]
    k_lstm_layer<<<lgrid, 256, 0, stream>>>(h1b[q], H_, H_ / 32, h2b[p],
                                            Wcat2, KPAD23, bias2, c2, h2b[q]);
    // layer 3: A = [h2_new | h3_prev]
    k_lstm_layer<<<lgrid, 256, 0, stream>>>(h2b[q], H_, H_ / 32, h3b[p],
                                            Wcat3, KPAD23, bias3, c3, h3b[q]);
    // decoder + self-feed write-back
    k_decoder<<<22, 256, 0, stream>>>(h3b[q], WdecB, bdec, out, x1, t);
  }
}
